// SelfAttention_90864328114913
// MI455X (gfx1250) — matrix operand order repaired and checked by reading, not compiled, not run on hardware
//
#include <hip/hip_runtime.h>
#include <hip/hip_bf16.h>

// ---------------------------------------------------------------------------
// CDNA5 (gfx1250) self-attention: bf16 WMMA GEMMs (TDM-staged) + flash attn.
// ---------------------------------------------------------------------------

typedef __attribute__((ext_vector_type(16))) __bf16 v16bf;
typedef __attribute__((ext_vector_type(8)))  float  v8f;
typedef __attribute__((ext_vector_type(4)))  unsigned int v4u;
typedef __attribute__((ext_vector_type(8)))  int  v8i;
typedef __attribute__((ext_vector_type(4)))  int  v4i;

#define D_MODEL 1024
#define N_HEAD  16
#define HEAD_DIM 64
#define SEQ_T   2048
#define BATCH   4
#define MROWS   (BATCH * SEQ_T)   // 8192
#define LDS_STRIDE 40             // shorts/row: 16dw tile row + 4dw TDM pad

__device__ __forceinline__ unsigned short f2bf(float f) {
    union { float f; unsigned u; } x; x.f = f;
    unsigned r = x.u + 0x7FFFu + ((x.u >> 16) & 1u);   // RNE
    return (unsigned short)(r >> 16);
}
__device__ __forceinline__ float bf2f(unsigned short h) {
    union { unsigned u; float f; } x; x.u = ((unsigned)h) << 16;
    return x.f;
}

union FragU { float4 q[2]; v16bf v; };

__device__ __forceinline__ v16bf load_frag(const unsigned short* p, int off_lo, int off_hi) {
    FragU u;
    u.q[0] = *reinterpret_cast<const float4*>(p + off_lo);
    u.q[1] = *reinterpret_cast<const float4*>(p + off_hi);
    return u.v;
}

__device__ __forceinline__ v8f wmma_bf16(v16bf a, v16bf b, v8f c) {
    return __builtin_amdgcn_wmma_f32_16x16x32_bf16(false, a, false, b, (short)0, c,
                                                   false, false);
}

// ---------------------------------------------------------------------------
// TDM: DMA a 2-D bf16 tile (tile_w x tile_h, row stride tensor_w) into LDS.
// Pads 4 DWORDs after every 16 DWORDs stored -> LDS row stride = 40 shorts.
// D# layout per CDNA5 ISA ch.8 (group0: count/lds/global/type; group1: dims).
// 6-arg builtin (clang-23 lane): groups 2/3 + aux group unused for 2-D tiles.
// ---------------------------------------------------------------------------
__device__ __forceinline__ void tdm_load_2d(unsigned lds_byte_off,
                                            const unsigned short* gptr,
                                            unsigned tensor_w, unsigned tensor_h,
                                            unsigned tile_w, unsigned tile_h) {
    unsigned long long ga = (unsigned long long)(uintptr_t)gptr;
    v4u g0;
    g0.x = 1u;                                   // count=1, is_restore=0
    g0.y = lds_byte_off;                         // lds_addr
    g0.z = (unsigned)ga;                         // global_addr[31:0]
    g0.w = (unsigned)(ga >> 32) | (2u << 30);    // global_addr[56:32] | type=2
    v8i g1;
    g1[0] = (1 << 16)          // data_size = 2 bytes
          | (1 << 20)          // pad_enable
          | (3 << 22)          // pad_interval: 16 DWORDs
          | (3 << 25);         // pad_amount:    4 DWORDs
    g1[1] = (int)(tensor_w << 16);                          // tensor_dim0 lo
    g1[2] = (int)((tensor_w >> 16) | (tensor_h << 16));     // dim0 hi | dim1 lo
    g1[3] = (int)((tensor_h >> 16) | (tile_w << 16));       // dim1 hi | tile_dim0
    g1[4] = (int)tile_h;                                    // tile_dim1 (dim2=0)
    g1[5] = (int)tensor_w;                                  // tensor_dim0_stride lo
    g1[6] = 0;
    g1[7] = 0;
    v4i z4 = {0, 0, 0, 0};
    v8i z8 = {0, 0, 0, 0, 0, 0, 0, 0};
    __builtin_amdgcn_tensor_load_to_lds(g0, g1, z4, z4, z8, 0);
}

// Low 32 bits of a generic LDS-aperture address are the LDS byte offset.
__device__ __forceinline__ unsigned lds_off(const void* p) {
    return (unsigned)(uintptr_t)p;
}

// ---------------------------------------------------------------------------
// bf16 GEMM: C[M,N] = A[M,K] * B[N,K]^T.  Block tile 128x128, 8 waves,
// wave tile 32x64 (8 WMMA / K-step).  TDM double-buffered LDS staging.
// ---------------------------------------------------------------------------
template <bool OUT_BF16>
__global__ __launch_bounds__(256) void gemm_wmma(const unsigned short* __restrict__ A,
                                                 const unsigned short* __restrict__ B,
                                                 void* __restrict__ Cv,
                                                 int M, int N, int K) {
    __shared__ unsigned short lA[2][128 * LDS_STRIDE];
    __shared__ unsigned short lB[2][128 * LDS_STRIDE];

    const int tid  = threadIdx.x;
    const int wave = tid >> 5;
    const int lane = tid & 31;
    const int m    = lane & 15;
    const int half = lane >> 4;
    const int wm   = wave & 3;    // 32-row slice
    const int wn   = wave >> 2;   // 64-col slice

    const unsigned short* Ab = A + (long)(blockIdx.y * 128) * K;
    const unsigned short* Bb = B + (long)(blockIdx.x * 128) * K;

    v8f acc[2][4];
#pragma unroll
    for (int s = 0; s < 2; ++s)
#pragma unroll
        for (int t = 0; t < 4; ++t) acc[s][t] = (v8f){0,0,0,0,0,0,0,0};

    const bool issuer = (wave == 0);
    if (issuer) {                             // prologue: stage buf 0
        tdm_load_2d(lds_off(&lA[0][0]), Ab, K, 128, 32, 128);
        tdm_load_2d(lds_off(&lB[0][0]), Bb, K, 128, 32, 128);
    }

    int buf = 0;
    for (int kt = 0; kt < K; kt += 32) {
        if (issuer) {
            if (kt + 32 < K) {                // prefetch next pair via DMA
                tdm_load_2d(lds_off(&lA[buf ^ 1][0]), Ab + kt + 32, K, 128, 32, 128);
                tdm_load_2d(lds_off(&lB[buf ^ 1][0]), Bb + kt + 32, K, 128, 32, 128);
                __builtin_amdgcn_s_wait_tensorcnt(2);   // current pair landed
            } else {
                __builtin_amdgcn_s_wait_tensorcnt(0);
            }
        }
        __syncthreads();                      // publish buf to all waves

        const unsigned short* la = lA[buf];
        const unsigned short* lb = lB[buf];
        v16bf af[2], bfr[4];
#pragma unroll
        for (int s = 0; s < 2; ++s) {
            const int ab = (wm * 32 + s * 16 + m) * LDS_STRIDE;
            af[s] = load_frag(la, ab + half * 8, ab + 16 + half * 8);
        }
#pragma unroll
        for (int t = 0; t < 4; ++t) {
            const int bb = (wn * 64 + t * 16 + m) * LDS_STRIDE + half * 8;
            bfr[t] = load_frag(lb, bb, bb + 16);
        }
#pragma unroll
        for (int s = 0; s < 2; ++s)
#pragma unroll
            for (int t = 0; t < 4; ++t)
                acc[s][t] = wmma_bf16(af[s], bfr[t], acc[s][t]);

        __syncthreads();                      // done reading before DMA reuse
        buf ^= 1;
    }

#pragma unroll
    for (int s = 0; s < 2; ++s) {
        const int row0 = blockIdx.y * 128 + wm * 32 + s * 16 + half * 8;
#pragma unroll
        for (int t = 0; t < 4; ++t) {
            const int col = blockIdx.x * 128 + wn * 64 + t * 16 + m;
#pragma unroll
            for (int r = 0; r < 8; ++r) {
                const long i0 = (long)(row0 + r) * N + col;
                if (OUT_BF16) ((unsigned short*)Cv)[i0] = f2bf(acc[s][t][r]);
                else          ((float*)Cv)[i0] = acc[s][t][r];
            }
        }
    }
}

// ---------------------------------------------------------------------------
// fp32 -> bf16 bulk convert (8 elements/thread).
// ---------------------------------------------------------------------------
__global__ __launch_bounds__(256) void to_bf16(const float* __restrict__ src,
                                               unsigned short* __restrict__ dst) {
    const long i = ((long)blockIdx.x * 256 + threadIdx.x) * 8;
    float4 a = *reinterpret_cast<const float4*>(src + i);
    float4 b = *reinterpret_cast<const float4*>(src + i + 4);
    unsigned short t[8] = { f2bf(a.x), f2bf(a.y), f2bf(a.z), f2bf(a.w),
                            f2bf(b.x), f2bf(b.y), f2bf(b.z), f2bf(b.w) };
    *reinterpret_cast<uint4*>(dst + i) = *reinterpret_cast<const uint4*>(t);
}

// ---------------------------------------------------------------------------
// RoPE (interleaved pairs -> concatenated halves); folds 1/sqrt(d) into Q.
// ---------------------------------------------------------------------------
__global__ __launch_bounds__(256) void rope_kernel(const unsigned short* __restrict__ Qr,
                                                   const unsigned short* __restrict__ Kr,
                                                   unsigned short* __restrict__ Qo,
                                                   unsigned short* __restrict__ Ko) {
    const unsigned idx = blockIdx.x * 256u + threadIdx.x;  // B*T*H*32 threads
    const int i = idx & 31;
    const int h = (idx >> 5) & 15;
    const int t = (idx >> 9) & 2047;
    const int b = idx >> 20;
    const long row = ((long)(b * SEQ_T + t)) * D_MODEL + h * HEAD_DIM;

    const float inv_freq = __expf(-0.28782313662425574f * (float)i); // 1e4^(-i/32)
    const float ang = (float)t * inv_freq;
    const float c = __cosf(ang), s = __sinf(ang);

    const float q1 = bf2f(Qr[row + 2 * i]), q2 = bf2f(Qr[row + 2 * i + 1]);
    const float k1 = bf2f(Kr[row + 2 * i]), k2 = bf2f(Kr[row + 2 * i + 1]);

    Qo[row + i]      = f2bf(0.125f * (q1 * c - q2 * s));
    Qo[row + 32 + i] = f2bf(0.125f * (q1 * s + q2 * c));
    Ko[row + i]      = f2bf(k1 * c - k2 * s);
    Ko[row + 32 + i] = f2bf(k1 * s + k2 * c);
}

// ---------------------------------------------------------------------------
// Flash attention: block = (b,h) x 128 query rows; 8 waves x 16 rows.
// Per 32-key chunk per wave: 4 WMMA (S) + 4 WMMA (P*V) + online softmax.
// ---------------------------------------------------------------------------
__global__ __launch_bounds__(256) void attn_kernel(const unsigned short* __restrict__ Q,
                                                   const unsigned short* __restrict__ K,
                                                   const unsigned short* __restrict__ V,
                                                   unsigned short* __restrict__ O) {
    __shared__ unsigned short ldsV[64 * LDS_STRIDE];        // V tile [d][key]
    __shared__ unsigned short ldsP[8][16 * LDS_STRIDE];     // per-wave P tile

    const int tid  = threadIdx.x;
    const int wave = tid >> 5;
    const int lane = tid & 31;
    const int m    = lane & 15;
    const int half = lane >> 4;

    const int b  = blockIdx.x >> 4;
    const int h  = blockIdx.x & 15;
    const int q0 = blockIdx.y * 128 + wave * 16;

    const long qbase = ((long)(b * SEQ_T + q0 + m)) * D_MODEL + h * HEAD_DIM;
    v16bf qfrag[2];
#pragma unroll
    for (int ks = 0; ks < 2; ++ks)
        qfrag[ks] = load_frag(Q, qbase + ks * 32 + half * 8,
                                 qbase + ks * 32 + 16 + half * 8);

    v8f o0 = {0,0,0,0,0,0,0,0}, o1 = o0, o2 = o0, o3 = o0;
    float mrow[8], lrow[8];
#pragma unroll
    for (int r = 0; r < 8; ++r) { mrow[r] = -1e30f; lrow[r] = 0.f; }

    const int vkey = tid & 31;
    const int vd0  = (tid >> 5) * 8;

    for (int kc = 0; kc < SEQ_T; kc += 32) {
        __syncthreads();

        {   // cooperative V tile load, transposed into [d][key]
            const unsigned short* vp =
                V + ((long)(b * SEQ_T + kc + vkey)) * D_MODEL + h * HEAD_DIM + vd0;
            uint4 raw = *reinterpret_cast<const uint4*>(vp);
            const unsigned short* e = reinterpret_cast<const unsigned short*>(&raw);
#pragma unroll
            for (int j = 0; j < 8; ++j)
                ldsV[(vd0 + j) * LDS_STRIDE + vkey] = e[j];
        }

        // S = Q * K^T over 32 keys (two 16x16 tiles)
        v8f s0 = {0,0,0,0,0,0,0,0}, s1 = s0;
#pragma unroll
        for (int ks = 0; ks < 2; ++ks) {
            const long kb0 = ((long)(b * SEQ_T + kc + m)) * D_MODEL
                             + h * HEAD_DIM + ks * 32 + half * 8;
            const long kb1 = ((long)(b * SEQ_T + kc + 16 + m)) * D_MODEL
                             + h * HEAD_DIM + ks * 32 + half * 8;
            v16bf kf0 = load_frag(K, kb0, kb0 + 16);
            v16bf kf1 = load_frag(K, kb1, kb1 + 16);
            s0 = wmma_bf16(qfrag[ks], kf0, s0);
            s1 = wmma_bf16(qfrag[ks], kf1, s1);
        }

        // online softmax, row stats across 16-lane groups
        float p0[8], p1[8];
#pragma unroll
        for (int r = 0; r < 8; ++r) {
            float t = fmaxf(s0[r], s1[r]);
            t = fmaxf(t, __shfl_xor(t, 1, 32));
            t = fmaxf(t, __shfl_xor(t, 2, 32));
            t = fmaxf(t, __shfl_xor(t, 4, 32));
            t = fmaxf(t, __shfl_xor(t, 8, 32));
            const float mn = fmaxf(mrow[r], t);
            const float alpha = __expf(mrow[r] - mn);
            mrow[r] = mn;
            p0[r] = __expf(s0[r] - mn);
            p1[r] = __expf(s1[r] - mn);
            float rs = p0[r] + p1[r];
            rs += __shfl_xor(rs, 1, 32);
            rs += __shfl_xor(rs, 2, 32);
            rs += __shfl_xor(rs, 4, 32);
            rs += __shfl_xor(rs, 8, 32);
            lrow[r] = lrow[r] * alpha + rs;
            o0[r] *= alpha; o1[r] *= alpha; o2[r] *= alpha; o3[r] *= alpha;
        }

        // re-stripe P into A-fragment layout via wave-private LDS
        unsigned short* pw = ldsP[wave];
#pragma unroll
        for (int r = 0; r < 8; ++r) {
            const int row = half * 8 + r;
            pw[row * LDS_STRIDE + m]      = f2bf(p0[r]);
            pw[row * LDS_STRIDE + 16 + m] = f2bf(p1[r]);
        }
        __syncthreads();

        // O += P(16x32) * V(32x64)
        const int pb = m * LDS_STRIDE;
        v16bf pfrag = load_frag(pw, pb + half * 8, pb + 16 + half * 8);
#pragma unroll
        for (int ds = 0; ds < 4; ++ds) {
            const int vb = (ds * 16 + m) * LDS_STRIDE + half * 8;
            v16bf vfrag = load_frag(ldsV, vb, vb + 16);
            v8f* oacc = (ds == 0) ? &o0 : (ds == 1) ? &o1 : (ds == 2) ? &o2 : &o3;
            *oacc = wmma_bf16(pfrag, vfrag, *oacc);
        }
    }

#pragma unroll
    for (int r = 0; r < 8; ++r) {
        const float inv = 1.0f / lrow[r];
        const long row = ((long)(b * SEQ_T + q0 + half * 8 + r)) * D_MODEL
                         + h * HEAD_DIM + m;
        O[row]      = f2bf(o0[r] * inv);
        O[row + 16] = f2bf(o1[r] * inv);
        O[row + 32] = f2bf(o2[r] * inv);
        O[row + 48] = f2bf(o3[r] * inv);
    }
}

// ---------------------------------------------------------------------------
// Launch: x,Wq,Wk,Wv,Wo (fp32) -> d_out (fp32).
// ---------------------------------------------------------------------------
extern "C" void kernel_launch(void* const* d_in, const int* in_sizes, int n_in,
                              void* d_out, int out_size, void* d_ws, size_t ws_size,
                              hipStream_t stream) {
    const float* x  = (const float*)d_in[0];
    const float* Wq = (const float*)d_in[1];
    const float* Wk = (const float*)d_in[2];
    const float* Wv = (const float*)d_in[3];
    const float* Wo = (const float*)d_in[4];

    const size_t E = (size_t)MROWS * D_MODEL;          // 8.39M elems
    const size_t W = (size_t)D_MODEL * D_MODEL;        // 1.05M elems
    unsigned short* ws  = (unsigned short*)d_ws;
    unsigned short* xb  = ws;                          // bf16 copies
    unsigned short* Wqb = ws + E;
    unsigned short* Wkb = ws + E + W;
    unsigned short* Wvb = ws + E + 2 * W;
    unsigned short* Wob = ws + E + 3 * W;
    unsigned short* Qraw = ws + E + 4 * W;             // reused as attn_out
    unsigned short* Kraw = Qraw + E;
    unsigned short* Vb   = Qraw + 2 * E;
    unsigned short* Qr   = Qraw + 3 * E;
    unsigned short* Kr   = Qraw + 4 * E;
    unsigned short* AO   = Qraw;

    to_bf16<<<E / 2048, 256, 0, stream>>>(x,  xb);
    to_bf16<<<W / 2048, 256, 0, stream>>>(Wq, Wqb);
    to_bf16<<<W / 2048, 256, 0, stream>>>(Wk, Wkb);
    to_bf16<<<W / 2048, 256, 0, stream>>>(Wv, Wvb);
    to_bf16<<<W / 2048, 256, 0, stream>>>(Wo, Wob);

    dim3 gemm_grid(D_MODEL / 128, MROWS / 128);        // (8, 64)
    gemm_wmma<true><<<gemm_grid, 256, 0, stream>>>(xb, Wqb, Qraw,
                                                   MROWS, D_MODEL, D_MODEL);
    gemm_wmma<true><<<gemm_grid, 256, 0, stream>>>(xb, Wkb, Kraw,
                                                   MROWS, D_MODEL, D_MODEL);
    gemm_wmma<true><<<gemm_grid, 256, 0, stream>>>(xb, Wvb, Vb,
                                                   MROWS, D_MODEL, D_MODEL);

    const int rope_threads = BATCH * SEQ_T * N_HEAD * (HEAD_DIM / 2);
    rope_kernel<<<rope_threads / 256, 256, 0, stream>>>(Qraw, Kraw, Qr, Kr);

    dim3 attn_grid(BATCH * N_HEAD, SEQ_T / 128);       // (64, 16)
    attn_kernel<<<attn_grid, 256, 0, stream>>>(Qr, Kr, Vb, AO);

    gemm_wmma<false><<<gemm_grid, 256, 0, stream>>>(AO, Wob, d_out,
                                                    MROWS, D_MODEL, D_MODEL);
}
